// MHSA_4681514352918
// MI455X (gfx1250) — matrix operand order repaired and checked by reading, not compiled, not run on hardware
//
#include <hip/hip_runtime.h>

// ---------------------------------------------------------------------------
// Types for WMMA fragments (CDNA5 / gfx1250, wave32)
// ---------------------------------------------------------------------------
typedef __attribute__((ext_vector_type(16))) __bf16        v16bf;
typedef __attribute__((ext_vector_type(8)))  float         v8f;
typedef __attribute__((ext_vector_type(4)))  unsigned int  u32x4;
typedef __attribute__((ext_vector_type(8)))  float         f32x8;

union V16A {                  // 32 bytes: one 16-element bf16 WMMA operand
    u32x4          q[2];
    unsigned short us[16];
    v16bf          bf;
};

static __device__ inline unsigned short f32_to_bf16(float f) {
    unsigned int u = __float_as_uint(f);
    u += 0x7FFFu + ((u >> 16) & 1u);      // round-to-nearest-even
    return (unsigned short)(u >> 16);
}

static __device__ inline v8f v8f_zero() {
    v8f r;
    for (int i = 0; i < 8; ++i) r[i] = 0.0f;
    return r;
}

static __device__ inline v8f wmma_bf16(v16bf a, v16bf b, v8f c) {
    return __builtin_amdgcn_wmma_f32_16x16x32_bf16(
        false, a, false, b, (short)0, c, false, false);
}

// ---------------------------------------------------------------------------
// Weight transpose + f32->bf16 convert:  Wt[n*K + k] = bf16(W[k*N + n])
// ---------------------------------------------------------------------------
__global__ void transpose_to_bf16(const float* __restrict__ W,
                                  unsigned short* __restrict__ Wt,
                                  int K, int N) {
    int idx = blockIdx.x * blockDim.x + threadIdx.x;
    if (idx < K * N) {
        int k = idx / N;
        int n = idx % N;
        Wt[(size_t)n * K + k] = f32_to_bf16(W[idx]);
    }
}

// ---------------------------------------------------------------------------
// Generic WMMA GEMM:  C[M,N] = A[M,K] @ Bt^T + bias
//   A:  row-major, f32 (AF32=true, converted in-register) or bf16
//   Bt: [N,K] bf16 row-major (i.e. B transposed) -> contiguous B-frag loads
//   OUTMODE 0: bf16 [M,N]   1: bf16 per-batch transposed [B][N][Lb]   2: f32 [M,N]
// Wave tile 16x64 (4 C-frags), 4 waves/WG -> 64x64 block tile.
// ---------------------------------------------------------------------------
template <bool AF32, int OUTMODE>
__global__ __launch_bounds__(128) void gemm_wmma_kernel(
    const void* __restrict__ Av, const unsigned short* __restrict__ Bt,
    const float* __restrict__ bias, void* __restrict__ outv,
    int M, int N, int K, int Lb) {
    const int lane = threadIdx.x & 31;
    const int wave = threadIdx.x >> 5;
    const int half = lane >> 4;
    const int ln   = lane & 15;
    const int m0   = blockIdx.x * 64 + wave * 16;
    const int n0   = blockIdx.y * 64;
    const int row  = m0 + ln;                    // A-fragment row for this lane

    v8f acc[4];
    for (int t = 0; t < 4; ++t) acc[t] = v8f_zero();

    for (int c = 0; c < K; c += 32) {
        V16A a;
        if (AF32) {
            const float* A = (const float*)Av + (size_t)row * K;
            f32x8 r0 = *(const f32x8*)(A + c + half * 8);
            f32x8 r1 = *(const f32x8*)(A + c + 16 + half * 8);
            for (int e = 0; e < 8; ++e) {
                a.us[e]     = f32_to_bf16(r0[e]);
                a.us[8 + e] = f32_to_bf16(r1[e]);
            }
        } else {
            const unsigned short* A = (const unsigned short*)Av + (size_t)row * K;
            a.q[0] = *(const u32x4*)(A + c + half * 8);
            a.q[1] = *(const u32x4*)(A + c + 16 + half * 8);
        }
        for (int t = 0; t < 4; ++t) {
            const int n = n0 + t * 16 + ln;
            const unsigned short* B = Bt + (size_t)n * K + c + half * 8;
            V16A b;
            b.q[0] = *(const u32x4*)(B);
            b.q[1] = *(const u32x4*)(B + 16);
            acc[t] = wmma_bf16(a.bf, b.bf, acc[t]);
        }
    }

    for (int t = 0; t < 4; ++t) {
        const int n  = n0 + t * 16 + ln;
        const float bb = bias ? bias[n] : 0.0f;
        for (int j = 0; j < 8; ++j) {
            const int r   = m0 + j + 8 * half;   // C-frag row
            const float v = acc[t][j] + bb;
            if (OUTMODE == 0) {
                ((unsigned short*)outv)[(size_t)r * N + n] = f32_to_bf16(v);
            } else if (OUTMODE == 1) {
                const int bidx = r / Lb, l = r % Lb;
                ((unsigned short*)outv)[((size_t)bidx * N + n) * Lb + l] =
                    f32_to_bf16(v);
            } else {
                ((float*)outv)[(size_t)r * N + n] = v;
            }
        }
    }
}

// ---------------------------------------------------------------------------
// Flash attention:  o = softmax(q k^T / 16) v      (L=4096, DK=256, per batch)
// q,k: [B][L][256] bf16 row-major;  vT: [B][256][L] bf16;  o: [B][L][256] bf16
// Block = 4 waves; wave owns 16 query rows x full DK.
// KV streamed in 32-row chunks, double-buffered in LDS via
// global_load_async_to_lds_b128 (ASYNCcnt). Fragment reads from LDS are
// software-pipelined one step ahead so ds latency overlaps WMMA issue.
// ---------------------------------------------------------------------------
__global__ __launch_bounds__(128) void attn_kernel(
    const unsigned short* __restrict__ q, const unsigned short* __restrict__ k,
    const unsigned short* __restrict__ vT, unsigned short* __restrict__ o) {
    constexpr int L = 4096, D = 256;
    constexpr int NCH = L / 32;                      // 128 kv chunks

    __shared__ __align__(128) unsigned short tileK[2][32 * 256]; // 2 x 16 KB
    __shared__ __align__(128) unsigned short tileV[2][256 * 32]; // 2 x 16 KB
    __shared__ __align__(128) unsigned short plds[4 * 16 * 32];  // P bounce

    const int tid   = threadIdx.x;
    const int lane  = tid & 31;
    const int wave  = tid >> 5;
    const int half  = lane >> 4;
    const int ln    = lane & 15;
    const int batch = blockIdx.y;
    const int qbase = blockIdx.x * 64 + wave * 16;
    unsigned short* myp = plds + wave * (16 * 32);

    const unsigned short* Kb = k  + (size_t)batch * L * D;
    const unsigned short* Vb = vT + (size_t)batch * D * L;

    // Cooperative async staging of one 32-kv chunk: K tile (32 x 256 bf16,
    // row-major) + V tile (256 x 32 bf16, row-major). 16-byte segments,
    // 2048 lane-transfers total -> 16 async instructions per wave.
    auto stage = [&](int kv0, int buf) {
        const unsigned ldsK = (unsigned)(uintptr_t)&tileK[buf][0];
        const unsigned ldsV = (unsigned)(uintptr_t)&tileV[buf][0];
        for (int it = 0; it < 8; ++it) {
            const int seg = it * 128 + tid;          // 0..1023
            const int row = seg >> 5, col = seg & 31;
            unsigned long long src =
                (unsigned long long)(uintptr_t)(Kb + (size_t)(kv0 + row) * D) +
                (unsigned)(col * 16);
            unsigned dst = ldsK + (unsigned)(row * 512 + col * 16);
            asm volatile("global_load_async_to_lds_b128 %0, %1, off"
                         :: "v"(dst), "v"(src) : "memory");
        }
        for (int it = 0; it < 8; ++it) {
            const int seg = it * 128 + tid;          // 0..1023
            const int row = seg >> 2, qq = seg & 3;
            unsigned long long src =
                (unsigned long long)(uintptr_t)(Vb + (size_t)row * L + kv0) +
                (unsigned)(qq * 16);
            unsigned dst = ldsV + (unsigned)(row * 64 + qq * 16);
            asm volatile("global_load_async_to_lds_b128 %0, %1, off"
                         :: "v"(dst), "v"(src) : "memory");
        }
    };

    // Load Q into 8 resident A-fragments (16 rows x 256 d)
    v16bf qf[8];
    {
        const unsigned short* Q = q + ((size_t)batch * L + qbase + ln) * D;
        for (int c = 0; c < 8; ++c) {
            V16A a;
            a.q[0] = *(const u32x4*)(Q + c * 32 + half * 8);
            a.q[1] = *(const u32x4*)(Q + c * 32 + 16 + half * 8);
            qf[c] = a.bf;
        }
    }

    v8f oacc[16];
    for (int f = 0; f < 16; ++f) oacc[f] = v8f_zero();
    float rowmax[8], rowsum[8];
    for (int j = 0; j < 8; ++j) { rowmax[j] = -1e30f; rowsum[j] = 0.0f; }

    // logits pre-scaled into log2 domain: log2(e) / sqrt(256)
    const float sc = 1.4426950408889634f / 16.0f;

    stage(0, 0);                                     // prefetch chunk 0

    for (int ich = 0; ich < NCH; ++ich) {
        const int kv0 = ich * 32;
        const int cur = ich & 1;

        if (ich + 1 < NCH) {
            stage(kv0 + 32, cur ^ 1);                // prefetch next chunk
            // async loads complete in order: <=16 outstanding means the 16
            // ops of the current chunk are done, next chunk still in flight
            asm volatile("s_wait_asynccnt 16" ::: "memory");
        } else {
            asm volatile("s_wait_asynccnt 0" ::: "memory");
        }
        __syncthreads();                             // buf[cur] visible to WG

        const unsigned short* tK = &tileK[cur][0];
        const unsigned short* tV = &tileV[cur][0];

        // ---- S = Q @ K^T, 16x32 logits tile; B-frags pipelined 1 ahead ----
        v8f s0 = v8f_zero(), s1 = v8f_zero();
        V16A b0c, b1c, b0n, b1n;
        {
            const unsigned short* B0 = tK + (size_t)ln * D + half * 8;
            b0c.q[0] = *(const u32x4*)(B0);
            b0c.q[1] = *(const u32x4*)(B0 + 16);
            const unsigned short* B1 = tK + (size_t)(16 + ln) * D + half * 8;
            b1c.q[0] = *(const u32x4*)(B1);
            b1c.q[1] = *(const u32x4*)(B1 + 16);
        }
#pragma unroll
        for (int c = 0; c < 8; ++c) {
            if (c + 1 < 8) {
                const unsigned short* B0 =
                    tK + (size_t)ln * D + (c + 1) * 32 + half * 8;
                b0n.q[0] = *(const u32x4*)(B0);
                b0n.q[1] = *(const u32x4*)(B0 + 16);
                const unsigned short* B1 =
                    tK + (size_t)(16 + ln) * D + (c + 1) * 32 + half * 8;
                b1n.q[0] = *(const u32x4*)(B1);
                b1n.q[1] = *(const u32x4*)(B1 + 16);
            }
            s0 = wmma_bf16(qf[c], b0c.bf, s0);
            s1 = wmma_bf16(qf[c], b1c.bf, s1);
            b0c = b0n;
            b1c = b1n;
        }

        // issue first V fragment early; softmax VALU below hides its latency
        V16A bvC, bvN;
        {
            const unsigned short* BV = tV + (size_t)ln * 32 + half * 8;
            bvC.q[0] = *(const u32x4*)(BV);
            bvC.q[1] = *(const u32x4*)(BV + 16);
        }

        // ---- online softmax (row reductions across each 16-lane half) ----
        float alpha[8];
#pragma unroll
        for (int j = 0; j < 8; ++j) {
            float t0 = s0[j] * sc, t1 = s1[j] * sc;
            float mx = fmaxf(t0, t1);
            mx = fmaxf(mx, __shfl_xor(mx, 1, 32));
            mx = fmaxf(mx, __shfl_xor(mx, 2, 32));
            mx = fmaxf(mx, __shfl_xor(mx, 4, 32));
            mx = fmaxf(mx, __shfl_xor(mx, 8, 32));
            const float mnew = fmaxf(rowmax[j], mx);
            const float a  = __builtin_amdgcn_exp2f(rowmax[j] - mnew);
            const float p0 = __builtin_amdgcn_exp2f(t0 - mnew);
            const float p1 = __builtin_amdgcn_exp2f(t1 - mnew);
            float rs = p0 + p1;
            rs += __shfl_xor(rs, 1, 32);
            rs += __shfl_xor(rs, 2, 32);
            rs += __shfl_xor(rs, 4, 32);
            rs += __shfl_xor(rs, 8, 32);
            rowsum[j] = rowsum[j] * a + rs;
            rowmax[j] = mnew;
            alpha[j]  = a;
            const int m = j + 8 * half;               // C-frag row
            myp[m * 32 + ln]      = f32_to_bf16(p0);  // P tile 16x32, row-major
            myp[m * 32 + 16 + ln] = f32_to_bf16(p1);
        }

        // ---- O *= alpha while the P stores drain ----
#pragma unroll
        for (int f = 0; f < 16; ++f)
            for (int j = 0; j < 8; ++j) oacc[f][j] *= alpha[j];

        asm volatile("s_wait_dscnt 0" ::: "memory");

        // ---- reload P in A-fragment layout from LDS ----
        V16A p;
        p.q[0] = *(const u32x4*)(myp + ln * 32 + half * 8);
        p.q[1] = *(const u32x4*)(myp + ln * 32 + 16 + half * 8);

        // ---- O += P @ V (V fragments pipelined 1 ahead) ----
#pragma unroll
        for (int f = 0; f < 16; ++f) {
            if (f + 1 < 16) {
                const unsigned short* BV =
                    tV + (size_t)((f + 1) * 16 + ln) * 32 + half * 8;
                bvN.q[0] = *(const u32x4*)(BV);
                bvN.q[1] = *(const u32x4*)(BV + 16);
            }
            oacc[f] = wmma_bf16(p.bf, bvC.bf, oacc[f]);
            bvC = bvN;
        }

        __syncthreads();   // all waves done with buf[cur] before it's restaged
    }

    // ---- normalize and store ----
    float inv[8];
    for (int j = 0; j < 8; ++j) inv[j] = __builtin_amdgcn_rcpf(rowsum[j]);
    for (int f = 0; f < 16; ++f)
        for (int j = 0; j < 8; ++j) {
            const int r = qbase + j + 8 * half;
            o[((size_t)batch * L + r) * D + f * 16 + ln] =
                f32_to_bf16(oacc[f][j] * inv[j]);
        }
}

// ---------------------------------------------------------------------------
// Host launcher
// ---------------------------------------------------------------------------
extern "C" void kernel_launch(void* const* d_in, const int* in_sizes, int n_in,
                              void* d_out, int out_size, void* d_ws, size_t ws_size,
                              hipStream_t stream) {
    (void)in_sizes; (void)n_in; (void)out_size; (void)ws_size;
    const float* x    = (const float*)d_in[0];
    const float* w_sr = (const float*)d_in[1];
    const float* b_sr = (const float*)d_in[2];
    const float* w_q  = (const float*)d_in[3];
    const float* b_q  = (const float*)d_in[4];
    const float* w_k  = (const float*)d_in[5];
    const float* b_k  = (const float*)d_in[6];
    const float* w_v  = (const float*)d_in[7];
    const float* b_v  = (const float*)d_in[8];
    const float* w_o  = (const float*)d_in[9];
    const float* b_o  = (const float*)d_in[10];

    constexpr int Nb = 8, WH = 16384, C = 256, R = 4, DK = 256;
    constexpr int Lr = WH / R;      // 4096
    constexpr int M  = Nb * Lr;     // 32768 rows
    constexpr int K1 = C * R;       // 1024

    char* ws = (char*)d_ws;
    size_t off = 0;
    auto carve = [&](size_t bytes) -> char* {
        char* p = ws + off;
        off += (bytes + 255) & ~(size_t)255;
        return p;
    };
    unsigned short* wsrT = (unsigned short*)carve((size_t)C  * K1 * 2);
    unsigned short* wqT  = (unsigned short*)carve((size_t)DK * C  * 2);
    unsigned short* wkT  = (unsigned short*)carve((size_t)DK * C  * 2);
    unsigned short* wvT  = (unsigned short*)carve((size_t)DK * C  * 2);
    unsigned short* woT  = (unsigned short*)carve((size_t)K1 * DK * 2);
    unsigned short* xr   = (unsigned short*)carve((size_t)M * C  * 2);
    unsigned short* qb   = (unsigned short*)carve((size_t)M * DK * 2);
    unsigned short* kb   = (unsigned short*)carve((size_t)M * DK * 2);
    unsigned short* vTb  = (unsigned short*)carve((size_t)M * DK * 2);
    unsigned short* ob   = (unsigned short*)carve((size_t)M * DK * 2);

    // Weight transposes (to [N][K] bf16)
    transpose_to_bf16<<<(K1 * C  + 255) / 256, 256, 0, stream>>>(w_sr, wsrT, K1, C);
    transpose_to_bf16<<<(C  * DK + 255) / 256, 256, 0, stream>>>(w_q,  wqT,  C,  DK);
    transpose_to_bf16<<<(C  * DK + 255) / 256, 256, 0, stream>>>(w_k,  wkT,  C,  DK);
    transpose_to_bf16<<<(C  * DK + 255) / 256, 256, 0, stream>>>(w_v,  wvT,  C,  DK);
    transpose_to_bf16<<<(DK * K1 + 255) / 256, 256, 0, stream>>>(w_o,  woT,  DK, K1);

    const dim3 blk(128);
    // xr = reshape(x) @ w_sr + b_sr          [32768,1024]x[1024,256] -> bf16
    gemm_wmma_kernel<true, 0><<<dim3(M / 64, C / 64), blk, 0, stream>>>(
        x, wsrT, b_sr, xr, M, C, K1, 0);
    // q / k / v projections                  [32768,256]x[256,256]
    gemm_wmma_kernel<false, 0><<<dim3(M / 64, DK / 64), blk, 0, stream>>>(
        xr, wqT, b_q, qb, M, DK, C, 0);
    gemm_wmma_kernel<false, 0><<<dim3(M / 64, DK / 64), blk, 0, stream>>>(
        xr, wkT, b_k, kb, M, DK, C, 0);
    gemm_wmma_kernel<false, 1><<<dim3(M / 64, DK / 64), blk, 0, stream>>>(
        xr, wvT, b_v, vTb, M, DK, C, Lr);       // v stored transposed per batch
    // attention
    attn_kernel<<<dim3(Lr / 64, Nb), blk, 0, stream>>>(qb, kb, vTb, ob);
    // out = o @ w_o + b_o -> f32 d_out       [32768,256]x[256,1024]
    gemm_wmma_kernel<false, 2><<<dim3(M / 64, K1 / 64), blk, 0, stream>>>(
        ob, woT, b_o, d_out, M, K1, DK, 0);
}
